// SelfAttentionHybrid_65481071406437
// MI455X (gfx1250) — matrix operand order repaired and checked by reading, not compiled, not run on hardware
//
#include <hip/hip_runtime.h>
#include <hip/hip_bf16.h>
#include <math.h>
#include <stdint.h>

// ---------------------------------------------------------------------------
// SelfAttentionHybrid for MI455X (gfx1250, wave32, WMMA + async LDS copies)
//   Xb = bf16(X); Rt = bf16(R^T); Et = bf16(E^T); Xt_b = bf16(X_b^T)
//   Q  = (Xb @ Rt^T) * (1/32)       [bf16]   (== X@R scaled)
//   K  =  Xb @ Et^T                 [bf16]   (== X@E)
//   per batch: S = Q @ K^T (fp32) ; P = softmax(S) [bf16]
//              out = sigmoid(P @ Xt^T)  (== P @ X_b)
// All GEMMs: v_wmma_f32_16x16x32_bf16, fp32 accumulation, double-buffered
// LDS staging via global_load_async_to_lds_b128 (ASYNCcnt) when available.
// ---------------------------------------------------------------------------

typedef __bf16 bf16;
typedef __attribute__((__ext_vector_type__(4)))  __bf16 v4bf;
typedef __attribute__((__ext_vector_type__(8)))  __bf16 v8bf;
typedef __attribute__((__ext_vector_type__(16))) __bf16 v16bf;
typedef __attribute__((__ext_vector_type__(4)))  float  v4f;
typedef __attribute__((__ext_vector_type__(8)))  float  v8f;
typedef int v4i __attribute__((vector_size(16)));

#define BM 128
#define BN 128
#define BK 64
#define LDSTR 72   // padded LDS row stride in halves (144B: 16B-aligned, spreads banks)

#if defined(__gfx1250__) && \
    __has_builtin(__builtin_amdgcn_global_load_async_to_lds_b128) && \
    __has_builtin(__builtin_amdgcn_s_wait_asynccnt)
#define HAS_ASYNC 1
#else
#define HAS_ASYNC 0
#endif

#define AS1 __attribute__((address_space(1)))
#define AS3 __attribute__((address_space(3)))

typedef AS1 v4i* gv4i_ptr;   // pointer to int4 in global (addrspace 1)
typedef AS3 v4i* lv4i_ptr;   // pointer to int4 in LDS    (addrspace 3)

__device__ __forceinline__ void copy16_g2l(const bf16* g, bf16* l)
{
#if HAS_ASYNC
    // per-lane 16B global -> LDS, tracked by ASYNCcnt (CDNA5 async path).
    // LDS generic pointer truncates to the 32-bit wave-relative offset
    // (flat aperture rule: the low 32 address bits are the LDS address).
    __builtin_amdgcn_global_load_async_to_lds_b128(
        (gv4i_ptr)(uintptr_t)g,
        (lv4i_ptr)(uint32_t)(uintptr_t)l, 0, 0);
#else
    *(v8bf*)l = *(const v8bf*)g;
#endif
}

__device__ __forceinline__ void wait_async()
{
#if HAS_ASYNC
    __builtin_amdgcn_s_wait_asynccnt(0);
#endif
}

// ---------------------------------------------------------------------------
// C[M,N] = act(alpha * A @ Bt^T)
//   A  : bf16 (M x K) row-major, lda
//   Bt : bf16 (N x K) row-major, ldb   (i.e. B pre-transposed, K-contiguous)
// Block: 256 threads (8 waves, wave32). Block tile 128x128, K-tile 64,
// double-buffered LDS; wave grid 4(M) x 2(N); wave tile 32x64 = 2x4 WMMA accs.
// ---------------------------------------------------------------------------
template<bool C_BF16, bool SIGMOID>
__global__ __launch_bounds__(256)
void gemm_wmma(const bf16* __restrict__ A, const bf16* __restrict__ Bt,
               void* __restrict__ Cp, int K, int lda, int ldb, int ldc,
               float alpha)
{
    __shared__ bf16 Alds[2][BM * LDSTR];   // [m][k]
    __shared__ bf16 Blds[2][BN * LDSTR];   // [n][k]

    const int tid  = threadIdx.x;
    const int m0   = blockIdx.y * BM;
    const int n0   = blockIdx.x * BN;

    const int wave   = tid >> 5;
    const int lane   = tid & 31;
    const int wm     = (wave & 3) * 32;
    const int wn     = (wave >> 2) * 64;
    const int laneM  = lane & 15;
    const int laneHi = lane >> 4;

    v8f acc[2][4];
    #pragma unroll
    for (int i = 0; i < 2; ++i)
        #pragma unroll
        for (int j = 0; j < 4; ++j)
            #pragma unroll
            for (int e = 0; e < 8; ++e)
                acc[i][j][e] = 0.0f;

    // stage one 128x64 A tile + 128x64 Bt tile: 2048 x 16B chunks, 8/thread
    auto stage = [&](int buf, int k0) {
        #pragma unroll
        for (int i = 0; i < 4; ++i) {
            const int c   = tid + i * 256;   // 0..1023
            const int row = c >> 3;          // 0..127
            const int cc  = (c & 7) * 8;     // halves along K
            copy16_g2l(A + (size_t)(m0 + row) * lda + k0 + cc,
                       &Alds[buf][row * LDSTR + cc]);
        }
        #pragma unroll
        for (int i = 0; i < 4; ++i) {
            const int c   = tid + i * 256;
            const int row = c >> 3;
            const int cc  = (c & 7) * 8;
            copy16_g2l(Bt + (size_t)(n0 + row) * ldb + k0 + cc,
                       &Blds[buf][row * LDSTR + cc]);
        }
    };

    const int nk = K / BK;
    stage(0, 0);
    wait_async();
    __syncthreads();

    for (int kt = 0; kt < nk; ++kt) {
        const int cur = kt & 1;
        if (kt + 1 < nk) stage(cur ^ 1, (kt + 1) * BK);   // prefetch next tile

        #pragma unroll
        for (int ks = 0; ks < 2; ++ks) {
            v16bf aF[2], bF[4];
            // A frag (ISA 16-bit A 16x32): lane(M) holds K = ks*32+laneHi*8+{0..7},
            // and +{16..23} -> two 16B chunks
            const int ka = ks * 32 + laneHi * 8;
            #pragma unroll
            for (int i = 0; i < 2; ++i) {
                const bf16* p = &Alds[cur][(wm + i * 16 + laneM) * LDSTR + ka];
                const v8bf lo = *(const v8bf*)p;
                const v8bf hi = *(const v8bf*)(p + 16);
                aF[i] = __builtin_shufflevector(lo, hi,
                        0,1,2,3,4,5,6,7,8,9,10,11,12,13,14,15);
            }
            // B frag: lane(N) holds K = ks*32+laneHi*8+{0..7}, and +{16..23}
            const int kb = ks * 32 + laneHi * 8;
            #pragma unroll
            for (int i = 0; i < 4; ++i) {
                const bf16* p = &Blds[cur][(wn + i * 16 + laneM) * LDSTR + kb];
                const v8bf lo = *(const v8bf*)p;
                const v8bf hi = *(const v8bf*)(p + 16);
                bF[i] = __builtin_shufflevector(lo, hi,
                        0,1,2,3,4,5,6,7,8,9,10,11,12,13,14,15);
            }
            #pragma unroll
            for (int mi = 0; mi < 2; ++mi)
                #pragma unroll
                for (int ni = 0; ni < 4; ++ni)
                    acc[mi][ni] = __builtin_amdgcn_wmma_f32_16x16x32_bf16(
                        false, aF[mi], false, bF[ni], (short)0, acc[mi][ni],
                        false, false);
        }
        wait_async();      // next buffer's async copies landed
        __syncthreads();   // everyone done reading current buffer
    }

    #pragma unroll
    for (int mi = 0; mi < 2; ++mi)
        #pragma unroll
        for (int ni = 0; ni < 4; ++ni)
            #pragma unroll
            for (int j = 0; j < 8; ++j) {
                const int m = m0 + wm + mi * 16 + laneHi * 8 + j;
                const int n = n0 + wn + ni * 16 + laneM;
                float v = acc[mi][ni][j] * alpha;
                if (SIGMOID) v = 1.0f / (1.0f + __expf(-v));
                if (C_BF16) ((bf16*)Cp)[(size_t)m * ldc + n] = (bf16)v;
                else        ((float*)Cp)[(size_t)m * ldc + n] = v;
            }
}

// ---------------------------------------------------------------------------
// fp32 -> bf16 elementwise (vec4)
__global__ __launch_bounds__(256)
void convert_f32_bf16(const float* __restrict__ in, bf16* __restrict__ out, int n)
{
    const int i = (blockIdx.x * 256 + threadIdx.x) * 4;
    if (i < n) {
        const v4f f = *(const v4f*)(in + i);
        v4bf w;
        #pragma unroll
        for (int j = 0; j < 4; ++j) w[j] = (bf16)f[j];
        *(v4bf*)(out + i) = w;
    }
}

// fp32 (rows x cols) -> bf16 transposed (cols x rows), 32x32 LDS tiles
__global__ __launch_bounds__(256)
void transpose_convert(const float* __restrict__ in, bf16* __restrict__ out,
                       int rows, int cols)
{
    __shared__ float t[32][33];
    const int bx = blockIdx.x * 32;          // col base
    const int by = blockIdx.y * 32;          // row base
    const int tx = threadIdx.x & 31;
    const int ty = threadIdx.x >> 5;         // 0..7
    #pragma unroll
    for (int i = 0; i < 32; i += 8)
        t[ty + i][tx] = in[(size_t)(by + ty + i) * cols + bx + tx];
    __syncthreads();
    #pragma unroll
    for (int i = 0; i < 32; i += 8)
        out[(size_t)(bx + ty + i) * rows + by + tx] = (bf16)t[tx][ty + i];
}

// numerically-stable row softmax: fp32 in (len 2048) -> bf16 out
__global__ __launch_bounds__(256)
void softmax_rows_2048(const float* __restrict__ S, bf16* __restrict__ P)
{
    __shared__ float red[256];
    const int tid = threadIdx.x;
    const float* row = S + (size_t)blockIdx.x * 2048;
    bf16*       prow = P + (size_t)blockIdx.x * 2048;

    float x[8];
    float mx = -3.0e38f;
    #pragma unroll
    for (int i = 0; i < 8; ++i) { x[i] = row[tid + i * 256]; mx = fmaxf(mx, x[i]); }

    red[tid] = mx;
    __syncthreads();
    for (int s = 128; s > 0; s >>= 1) {
        if (tid < s) red[tid] = fmaxf(red[tid], red[tid + s]);
        __syncthreads();
    }
    mx = red[0];
    __syncthreads();

    float sum = 0.0f;
    #pragma unroll
    for (int i = 0; i < 8; ++i) { x[i] = __expf(x[i] - mx); sum += x[i]; }

    red[tid] = sum;
    __syncthreads();
    for (int s = 128; s > 0; s >>= 1) {
        if (tid < s) red[tid] += red[tid + s];
        __syncthreads();
    }
    const float inv = 1.0f / red[0];
    #pragma unroll
    for (int i = 0; i < 8; ++i) prow[tid + i * 256] = (bf16)(x[i] * inv);
}

// ---------------------------------------------------------------------------
extern "C" void kernel_launch(void* const* d_in, const int* in_sizes, int n_in,
                              void* d_out, int out_size, void* d_ws, size_t ws_size,
                              hipStream_t stream)
{
    (void)in_sizes; (void)n_in; (void)out_size; (void)ws_size;

    const float* X = (const float*)d_in[0];   // (4, 2048, 1024) fp32
    const float* R = (const float*)d_in[1];   // (1024, 1024)    fp32
    const float* E = (const float*)d_in[2];   // (1024, 1024)    fp32
    float* out = (float*)d_out;               // (4, 2048, 1024) fp32

    const int B = 4, Sq = 2048, D = 1024;
    const int Mflat = B * Sq;                 // 8192

    // Workspace layout (92 MB peak):
    char* ws = (char*)d_ws;
    size_t off = 0;
    bf16*  Xb = (bf16*)(ws + off); off += (size_t)Mflat * D * 2;   // 16 MB
    bf16*  Xt = (bf16*)(ws + off); off += (size_t)Mflat * D * 2;   // 16 MB (per-batch transposed)
    bf16*  Rt = (bf16*)(ws + off); off += (size_t)D * D * 2;       //  2 MB
    bf16*  Et = (bf16*)(ws + off); off += (size_t)D * D * 2;       //  2 MB
    bf16*  Qb = (bf16*)(ws + off); off += (size_t)Mflat * D * 2;   // 16 MB
    bf16*  Kb = (bf16*)(ws + off); off += (size_t)Mflat * D * 2;   // 16 MB
    float* S  = (float*)(ws + off); off += (size_t)Sq * Sq * 4;    // 16 MB (reused/batch)
    bf16*  Pb = (bf16*)(ws + off); off += (size_t)Sq * Sq * 2;     //  8 MB (reused/batch)

    const dim3 blk(256);

    // 0) precision/layout prep (bandwidth-trivial vs 103 GFLOP of GEMM)
    convert_f32_bf16<<<(Mflat * D) / (256 * 4), blk, 0, stream>>>(X, Xb, Mflat * D);
    transpose_convert<<<dim3(D / 32, D / 32), blk, 0, stream>>>(R, Rt, D, D);
    transpose_convert<<<dim3(D / 32, D / 32), blk, 0, stream>>>(E, Et, D, D);
    for (int b = 0; b < B; ++b)
        transpose_convert<<<dim3(D / 32, Sq / 32), blk, 0, stream>>>(
            X + (size_t)b * Sq * D, Xt + (size_t)b * Sq * D, Sq, D);

    // 1) projections; 1/sqrt(1024)=1/32 folded into Q
    const dim3 gp(D / BN, Mflat / BM);        // (8, 64)
    gemm_wmma<true, false><<<gp, blk, 0, stream>>>(Xb, Rt, Qb, D, D, D, D, 1.0f / 32.0f);
    gemm_wmma<true, false><<<gp, blk, 0, stream>>>(Xb, Et, Kb, D, D, D, D, 1.0f);

    // 2) per batch: scores -> softmax(bf16) -> sigmoid(P @ X)
    const dim3 gs(Sq / BN, Sq / BM);          // (16, 16)
    const dim3 go(D  / BN, Sq / BM);          // (8, 16)
    for (int b = 0; b < B; ++b) {
        const bf16* Qbb = Qb + (size_t)b * Sq * D;
        const bf16* Kbb = Kb + (size_t)b * Sq * D;
        const bf16* Xtb = Xt + (size_t)b * Sq * D;
        float*      Ob  = out + (size_t)b * Sq * D;

        gemm_wmma<false, false><<<gs, blk, 0, stream>>>(Qbb, Kbb, S, D, D, D, Sq, 1.0f);
        softmax_rows_2048<<<Sq, blk, 0, stream>>>(S, Pb);
        gemm_wmma<false, true><<<go, blk, 0, stream>>>(Pb, Xtb, Ob, Sq, Sq, Sq, D, 1.0f);
    }
}
